// TinyModel_21749714387206
// MI455X (gfx1250) — compile-verified, matrix operand order checked by reading, not run on hardware
//
#include <hip/hip_runtime.h>

// ---------- types ----------
typedef __attribute__((ext_vector_type(16))) __bf16 v16bf;
typedef __attribute__((ext_vector_type(8)))  float  v8f;

#define EPSV 1e-5f
#define MB 4   // M-tiles register-blocked per wave (64 output rows / wave)

static inline int ceil_div(long a, long b) { return (int)((a + b - 1) / b); }

// ---------- helpers ----------
__device__ __forceinline__ void atomicMaxFloat(float* addr, float v) {
    if (v >= 0.0f) atomicMax((int*)addr, __float_as_int(v));
    else           atomicMin((unsigned int*)addr, __float_as_uint(v));
}

// Load 16 bf16 per lane in the CDNA5 16-bit A/B WMMA striping:
// elems 0..7 = K[base+0..7], elems 8..15 = K[base+16..23] (base already
// includes the +8 shift for lanes 16..31).
__device__ __forceinline__ v16bf ld_wmma16(const __bf16* __restrict__ p) {
    v16bf r;
#pragma unroll
    for (int i = 0; i < 8; ++i) { r[i] = p[i]; r[i + 8] = p[i + 16]; }
    return r;
}

// ---------- generic fills / converts ----------
__global__ void k_fill(float* __restrict__ p, float v, long n) {
    long i = (long)blockIdx.x * blockDim.x + threadIdx.x;
    long st = (long)gridDim.x * blockDim.x;
    for (; i < n; i += st) p[i] = v;
}

__global__ void k_f32_to_bf16(const float* __restrict__ s, __bf16* __restrict__ d, long n) {
    long i = (long)blockIdx.x * blockDim.x + threadIdx.x;
    long st = (long)gridDim.x * blockDim.x;
    for (; i < n; i += st) d[i] = (__bf16)s[i];
}

__global__ void k_init_rows(float* __restrict__ out, const float* __restrict__ bias, long n, int D) {
    long i = (long)blockIdx.x * blockDim.x + threadIdx.x;
    long st = (long)gridDim.x * blockDim.x;
    for (; i < n; i += st) out[i] = bias[i % D];
}

// ---------- WMMA GEMM: C[M,NN] = A[M,K](bf16) * Bw[NN,K](bf16)^T + bias ----------
// One wave per 64x16 output block (MB=4 register-blocked 16x16 WMMA tiles).
// Rows of A and C must be padded to a multiple of 64; K multiple of 32,
// NN multiple of 16. One B fragment feeds 4 WMMAs -> 4x less B traffic.
__global__ void k_gemm_bf16(const __bf16* __restrict__ A, const __bf16* __restrict__ Bw,
                            const float* __restrict__ bias, float* __restrict__ C,
                            int NN, int K) {
    int tiles_n = NN >> 4;
    int blk  = blockIdx.x;
    int tmb  = blk / tiles_n;          // super-tile (64 rows)
    int tn   = blk % tiles_n;
    int lane = threadIdx.x;
    int half = lane >> 4;              // 0: K{0-7,16-23}, 1: K{8-15,24-31}
    int r    = lane & 15;

    const __bf16* brow = Bw + (size_t)(tn * 16 + r) * K + half * 8;
    const __bf16* arow = A + (size_t)(tmb * 64 + r) * K + half * 8;
    const size_t rstep = (size_t)16 * K;

    v8f acc0 = {}, acc1 = {}, acc2 = {}, acc3 = {};
    for (int k0 = 0; k0 < K; k0 += 32) {
        if (k0 + 32 < K) {
            __builtin_prefetch(brow + k0 + 32, 0, 1);   // -> global_prefetch_b8
            __builtin_prefetch(arow + k0 + 32, 0, 1);
        }
        v16bf b  = ld_wmma16(brow + k0);
        v16bf a0 = ld_wmma16(arow + k0);
        v16bf a1 = ld_wmma16(arow + rstep + k0);
        v16bf a2 = ld_wmma16(arow + 2 * rstep + k0);
        v16bf a3 = ld_wmma16(arow + 3 * rstep + k0);
        acc0 = __builtin_amdgcn_wmma_f32_16x16x32_bf16(false, a0, false, b, (short)0, acc0, false, false);
        acc1 = __builtin_amdgcn_wmma_f32_16x16x32_bf16(false, a1, false, b, (short)0, acc1, false, false);
        acc2 = __builtin_amdgcn_wmma_f32_16x16x32_bf16(false, a2, false, b, (short)0, acc2, false, false);
        acc3 = __builtin_amdgcn_wmma_f32_16x16x32_bf16(false, a3, false, b, (short)0, acc3, false, false);
    }
    int n = tn * 16 + r;
    float bv = bias ? bias[n] : 0.0f;
    float* crow = C + (size_t)(tmb * 64 + half * 8) * NN + n;
#pragma unroll
    for (int v = 0; v < 8; ++v) {      // VGPR v: lanes0-15 M=v, lanes16-31 M=8+v
        crow[(size_t)v * NN]                     = acc0[v] + bv;
        crow[(size_t)(v + 16) * NN]              = acc1[v] + bv;
        crow[(size_t)(v + 32) * NN]              = acc2[v] + bv;
        crow[(size_t)(v + 48) * NN]              = acc3[v] + bv;
    }
}

// ---------- attention scalar projections: as[i]=h[i].a_src, ad[i]=h[i].a_dst ----------
__global__ void k_att(const float* __restrict__ h, const float* __restrict__ aws,
                      const float* __restrict__ awd, float* __restrict__ asn,
                      float* __restrict__ adn, int N, int D) {
    int node = (int)(((long)blockIdx.x * blockDim.x + threadIdx.x) >> 5);
    int lane = threadIdx.x & 31;
    if (node >= N) return;
    const float* row = h + (size_t)node * D;
    float s = 0.f, d = 0.f;
    for (int f = lane; f < D; f += 32) { float v = row[f]; s += v * aws[f]; d += v * awd[f]; }
#pragma unroll
    for (int off = 16; off > 0; off >>= 1) { s += __shfl_xor(s, off, 32); d += __shfl_xor(d, off, 32); }
    if (lane == 0) { asn[node] = s; adn[node] = d; }
}

// ---------- edge phase ----------
__global__ void k_edge_scores(const int* __restrict__ srcA, const int* __restrict__ dstA,
                              const float* __restrict__ asn, const float* __restrict__ adn,
                              float* __restrict__ e, float* __restrict__ m, int E, int Etot) {
    int i = blockIdx.x * blockDim.x + threadIdx.x;
    if (i >= Etot) return;
    int s, d;
    if (i < E) { s = srcA[i]; d = dstA[i]; } else { s = d = i - E; }
    float v = asn[s] + adn[d];
    v = (v > 0.f) ? v : 0.2f * v;            // leaky_relu(0.2)
    e[i] = v;
    atomicMaxFloat(&m[d], v);
}

__global__ void k_edge_exp(const int* __restrict__ dstA, float* __restrict__ e,
                           const float* __restrict__ m, float* __restrict__ den,
                           int E, int Etot) {
    int i = blockIdx.x * blockDim.x + threadIdx.x;
    if (i >= Etot) return;
    int d = (i < E) ? dstA[i] : (i - E);
    float ex = __expf(e[i] - m[d]);
    e[i] = ex;
    atomicAdd(&den[d], ex);
}

// wave per edge: out[dst] += alpha * h[src]
__global__ void k_edge_scatter(const int* __restrict__ srcA, const int* __restrict__ dstA,
                               const float* __restrict__ ex, const float* __restrict__ den,
                               const float* __restrict__ h, float* __restrict__ out,
                               int E, int Etot, int D) {
    int eid = (int)(((long)blockIdx.x * blockDim.x + threadIdx.x) >> 5);
    int lane = threadIdx.x & 31;
    if (eid >= Etot) return;
    int s, d;
    if (eid < E) { s = srcA[eid]; d = dstA[eid]; } else { s = d = eid - E; }
    float alpha = ex[eid] / den[d];
    const float* hr = h + (size_t)s * D;
    float* orow = out + (size_t)d * D;
    for (int f = lane; f < D; f += 32) atomicAdd(&orow[f], alpha * hr[f]);
}

// ---------- column reductions (feature-wise over rows) ----------
#define GN_CHUNK 64
__global__ void k_colsum(const float* __restrict__ h, float* __restrict__ sum, int N, int D) {
    int f = threadIdx.x;                     // blockDim.x == D
    int i0 = blockIdx.x * GN_CHUNK;
    int i1 = i0 + GN_CHUNK; if (i1 > N) i1 = N;
    float acc = 0.f;
    for (int i = i0; i < i1; ++i) acc += h[(size_t)i * D + f];
    atomicAdd(&sum[f], acc);
}

// h <- h - mu*ms (in place), accumulate centered squares
__global__ void k_gn_center_sq(float* __restrict__ h, const float* __restrict__ sum,
                               const float* __restrict__ ms, float* __restrict__ sq,
                               int N, int D) {
    int f = threadIdx.x;
    float mu = sum[f] / (float)N * ms[f];
    int i0 = blockIdx.x * GN_CHUNK;
    int i1 = i0 + GN_CHUNK; if (i1 > N) i1 = N;
    float acc = 0.f;
    for (int i = i0; i < i1; ++i) {
        float t = h[(size_t)i * D + f] - mu;
        h[(size_t)i * D + f] = t;
        acc += t * t;
    }
    atomicAdd(&sq[f], acc);
}

__global__ void k_gn_finalize(const float* __restrict__ h, const float* __restrict__ sq,
                              const float* __restrict__ w, const float* __restrict__ b,
                              __bf16* __restrict__ hb, long n, int D, int N) {
    long i = (long)blockIdx.x * blockDim.x + threadIdx.x;
    long st = (long)gridDim.x * blockDim.x;
    for (; i < n; i += st) {
        int f = (int)(i % D);
        float inv = __frsqrt_rn(sq[f] / (float)N + EPSV);
        float v = h[i] * inv * w[f] + b[f];
        hb[i] = (__bf16)(v > 0.f ? v : 0.f);  // fused ReLU
    }
}

// ---------- head ----------
__global__ void k_gather_z(const int* __restrict__ x, const __bf16* __restrict__ hb,
                           __bf16* __restrict__ z, int B, int D) {
    long n = (long)B * 3 * D;
    long i = (long)blockIdx.x * blockDim.x + threadIdx.x;
    long st = (long)gridDim.x * blockDim.x;
    for (; i < n; i += st) {
        int q = (int)(i / (3 * D));
        int rem = (int)(i % (3 * D));
        int c = rem / D, f = rem % D;
        z[i] = hb[(size_t)x[q * 3 + c] * D + f];
    }
}

#define BN_CHUNK 32
__global__ void k_colsum2(const float* __restrict__ z, float* __restrict__ s,
                          float* __restrict__ ss, int B, int C) {
    int f = threadIdx.x;                     // blockDim.x == C (384)
    int i0 = blockIdx.x * BN_CHUNK;
    int i1 = i0 + BN_CHUNK; if (i1 > B) i1 = B;
    float a = 0.f, a2 = 0.f;
    for (int i = i0; i < i1; ++i) { float v = z[(size_t)i * C + f]; a += v; a2 += v * v; }
    atomicAdd(&s[f], a); atomicAdd(&ss[f], a2);
}

__global__ void k_bn_relu_bf16(const float* __restrict__ z, const float* __restrict__ s,
                               const float* __restrict__ ss, const float* __restrict__ w,
                               const float* __restrict__ b, __bf16* __restrict__ o,
                               int B, int C) {
    long n = (long)B * C;
    long i = (long)blockIdx.x * blockDim.x + threadIdx.x;
    long st = (long)gridDim.x * blockDim.x;
    for (; i < n; i += st) {
        int f = (int)(i % C);
        float mu = s[f] / (float)B;
        float var = ss[f] / (float)B - mu * mu;
        float v = (z[i] - mu) * __frsqrt_rn(var + EPSV) * w[f] + b[f];
        o[i] = (__bf16)(v > 0.f ? v : 0.f);
    }
}

__global__ void k_log_softmax(float* __restrict__ out, int C) {
    __shared__ float red[256];
    float* p = out + (size_t)blockIdx.x * C;
    int t = threadIdx.x;
    float mx = -INFINITY;
    for (int j = t; j < C; j += 256) mx = fmaxf(mx, p[j]);
    red[t] = mx; __syncthreads();
    for (int s = 128; s > 0; s >>= 1) { if (t < s) red[t] = fmaxf(red[t], red[t + s]); __syncthreads(); }
    mx = red[0]; __syncthreads();
    float sm = 0.f;
    for (int j = t; j < C; j += 256) sm += __expf(p[j] - mx);
    red[t] = sm; __syncthreads();
    for (int s = 128; s > 0; s >>= 1) { if (t < s) red[t] += red[t + s]; __syncthreads(); }
    float lse = mx + __logf(red[0]); __syncthreads();
    for (int j = t; j < C; j += 256) p[j] = p[j] - lse;
}

// ---------- orchestration ----------
extern "C" void kernel_launch(void* const* d_in, const int* in_sizes, int n_in,
                              void* d_out, int out_size, void* d_ws, size_t ws_size,
                              hipStream_t stream) {
    const int*   x    = (const int*)d_in[0];
    const int*   ei   = (const int*)d_in[1];
    const float* emb  = (const float*)d_in[2];
    const float* W1   = (const float*)d_in[3];
    const float* as1  = (const float*)d_in[4];
    const float* ad1  = (const float*)d_in[5];
    const float* b1   = (const float*)d_in[6];
    const float* g1w  = (const float*)d_in[7];
    const float* g1b  = (const float*)d_in[8];
    const float* g1m  = (const float*)d_in[9];
    const float* W2   = (const float*)d_in[10];
    const float* as2  = (const float*)d_in[11];
    const float* ad2  = (const float*)d_in[12];
    const float* b2   = (const float*)d_in[13];
    const float* g2w  = (const float*)d_in[14];
    const float* g2b  = (const float*)d_in[15];
    const float* g2m  = (const float*)d_in[16];
    const float* l1W  = (const float*)d_in[17];
    const float* l1b  = (const float*)d_in[18];
    const float* bnw  = (const float*)d_in[19];
    const float* bnb  = (const float*)d_in[20];
    const float* l2W  = (const float*)d_in[21];
    const float* l2b  = (const float*)d_in[22];

    const int D  = in_sizes[4];          // 128
    const int B  = in_sizes[0] / 3;      // 1024
    const int E  = in_sizes[1] / 2;      // 800000
    const int N  = in_sizes[22];         // 50000
    const int C3 = in_sizes[18];         // 384
    const int Etot = E + N;
    const int Npad = ((N + 63) / 64) * 64;   // 50048: GEMM M-dim padded to 64
    const int* srcA = ei;
    const int* dstA = ei + E;

    // bump allocator on d_ws
    char* wp = (char*)d_ws;
    auto alloc = [&](size_t bytes) { void* p = wp; wp += (bytes + 255) & ~(size_t)255; return p; };
    __bf16* embb  = (__bf16*)alloc((size_t)Npad * D * 2);  // padded rows (garbage tail ok)
    __bf16* W1b   = (__bf16*)alloc((size_t)D * D * 2);
    __bf16* W2b   = (__bf16*)alloc((size_t)D * D * 2);
    __bf16* l1Wb  = (__bf16*)alloc((size_t)C3 * C3 * 2);
    __bf16* l2Wb  = (__bf16*)alloc((size_t)N * C3 * 2);
    float*  hf    = (float*)alloc((size_t)Npad * D * 4);   // Wh (GEMM out, padded)
    float*  of    = (float*)alloc((size_t)N * D * 4);      // aggregated
    __bf16* hb    = (__bf16*)alloc((size_t)Npad * D * 2);  // normed activations (padded)
    float*  asn   = (float*)alloc((size_t)N * 4);
    float*  adn   = (float*)alloc((size_t)N * 4);
    float*  mseg  = (float*)alloc((size_t)N * 4);
    float*  den   = (float*)alloc((size_t)N * 4);
    float*  ebuf  = (float*)alloc((size_t)Etot * 4);
    __bf16* zb    = (__bf16*)alloc((size_t)B * C3 * 2);
    float*  z1    = (float*)alloc((size_t)B * C3 * 4);
    __bf16* z2b   = (__bf16*)alloc((size_t)B * C3 * 2);
    float*  gsum  = (float*)alloc((size_t)D * 4);
    float*  gsq   = (float*)alloc((size_t)D * 4);
    float*  bs    = (float*)alloc((size_t)C3 * 4);
    float*  bss   = (float*)alloc((size_t)C3 * 4);

    const int T = 256;
    long nd  = (long)N * D;
    long ndp = (long)Npad * D;

    // weight conversions (per call; padded tails left as-is, never consumed)
    k_f32_to_bf16<<<ceil_div(nd, T), T, 0, stream>>>(emb, embb, nd);
    k_f32_to_bf16<<<ceil_div((long)D * D, T), T, 0, stream>>>(W1, W1b, (long)D * D);
    k_f32_to_bf16<<<ceil_div((long)D * D, T), T, 0, stream>>>(W2, W2b, (long)D * D);
    k_f32_to_bf16<<<ceil_div((long)C3 * C3, T), T, 0, stream>>>(l1W, l1Wb, (long)C3 * C3);
    k_f32_to_bf16<<<ceil_div((long)N * C3, T), T, 0, stream>>>(l2W, l2Wb, (long)N * C3);

    const __bf16* layer_in = embb;
    for (int L = 0; L < 2; ++L) {
        const __bf16* Wb  = (L == 0) ? W1b : W2b;
        const float *aws = (L == 0) ? as1 : as2, *awd = (L == 0) ? ad1 : ad2;
        const float *bias = (L == 0) ? b1 : b2;
        const float *gw = (L == 0) ? g1w : g2w, *gb = (L == 0) ? g1b : g2b, *gm = (L == 0) ? g1m : g2m;

        // h = X @ W^T via register-blocked WMMA (64 rows / wave)
        k_gemm_bf16<<<(Npad / 64) * (D / 16), 32, 0, stream>>>(layer_in, Wb, nullptr, hf, D, D);
        // attention scalars
        k_att<<<ceil_div((long)N * 32, T), T, 0, stream>>>(hf, aws, awd, asn, adn, N, D);
        // segment softmax over dst
        k_fill<<<ceil_div((long)N, T), T, 0, stream>>>(mseg, -INFINITY, N);
        k_fill<<<ceil_div((long)N, T), T, 0, stream>>>(den, 0.f, N);
        k_init_rows<<<ceil_div(nd, T), T, 0, stream>>>(of, bias, nd, D);
        k_edge_scores<<<ceil_div((long)Etot, T), T, 0, stream>>>(srcA, dstA, asn, adn, ebuf, mseg, E, Etot);
        k_edge_exp<<<ceil_div((long)Etot, T), T, 0, stream>>>(dstA, ebuf, mseg, den, E, Etot);
        k_edge_scatter<<<ceil_div((long)Etot * 32, T), T, 0, stream>>>(srcA, dstA, ebuf, den, hf, of, E, Etot, D);
        // GraphNorm + ReLU -> bf16
        k_fill<<<1, D, 0, stream>>>(gsum, 0.f, D);
        k_fill<<<1, D, 0, stream>>>(gsq, 0.f, D);
        k_colsum<<<ceil_div(N, GN_CHUNK), D, 0, stream>>>(of, gsum, N, D);
        k_gn_center_sq<<<ceil_div(N, GN_CHUNK), D, 0, stream>>>(of, gsum, gm, gsq, N, D);
        k_gn_finalize<<<ceil_div(nd, T), T, 0, stream>>>(of, gsq, gw, gb, hb, nd, D, N);
        layer_in = hb;
    }

    // head
    k_gather_z<<<ceil_div((long)B * C3, T), T, 0, stream>>>(x, hb, zb, B, D);
    k_gemm_bf16<<<(B / 64) * (C3 / 16), 32, 0, stream>>>(zb, l1Wb, l1b, z1, C3, C3);
    k_fill<<<ceil_div((long)C3, T), T, 0, stream>>>(bs, 0.f, C3);
    k_fill<<<ceil_div((long)C3, T), T, 0, stream>>>(bss, 0.f, C3);
    k_colsum2<<<ceil_div(B, BN_CHUNK), C3, 0, stream>>>(z1, bs, bss, B, C3);
    k_bn_relu_bf16<<<ceil_div((long)B * C3, T), T, 0, stream>>>(z1, bs, bss, bnw, bnb, z2b, B, C3);
    // classifier GEMM straight into d_out, then in-place log_softmax
    float* logits = (float*)d_out;
    k_gemm_bf16<<<(B / 64) * (N / 16), 32, 0, stream>>>(z2b, l2Wb, l2b, logits, N, C3);
    k_log_softmax<<<B, 256, 0, stream>>>(logits, N);
}
